// ContrastiveLoss_29059748725105
// MI455X (gfx1250) — compile-verified, matrix operand order checked by reading, not run on hardware
//
#include <hip/hip_runtime.h>
#include <hip/hip_bf16.h>

// ---------------------------------------------------------------------------
// Problem constants (from reference)
// ---------------------------------------------------------------------------
#define BB      4
#define CC      128
#define HH      256
#define WW      256
#define HWSZ    (HH * WW)            // 65536
#define CHW     (CC * HWSZ)          // 8388608
#define N_ANCH  8192
#define N_POS   50000
#define N_NEG   15000
#define N_TILES 938                  // ceil(15000/16)
#define N_PAD   (N_TILES * 16)       // 15008
#define PAD_EXTRA ((float)(N_PAD - N_NEG))   // 8 zero columns -> exp(0)=1 each
#define N_CHUNK 16                   // column-tile chunks per row-tile
#define INV_TEMP 10.0f

typedef __attribute__((ext_vector_type(16))) _Float16 v16h;
typedef __attribute__((ext_vector_type(8)))  _Float16 v8h;
typedef __attribute__((ext_vector_type(8)))  float    v8f;

// f(p, c) = feat[b][c][h][w] with p = b*H*W + h*W + w  (HW = 65536 = 2^16)
__device__ __forceinline__ float gatherF(const float* __restrict__ feat, int p, int c) {
    const int b   = p >> 16;
    const int rem = p & 0xFFFF;
    return feat[(size_t)b * CHW + (size_t)c * HWSZ + rem];
}

// ---------------------------------------------------------------------------
// 1) Zero accumulators (pos_sum[128], denom[8192])
// ---------------------------------------------------------------------------
__global__ void init_kernel(float* __restrict__ pos_sum, float* __restrict__ denom) {
    const int i = blockIdx.x * blockDim.x + threadIdx.x;
    if (i < CC)     pos_sum[i] = 0.0f;
    if (i < N_ANCH) denom[i]   = 0.0f;
}

// ---------------------------------------------------------------------------
// 2) Gather + convert: anchors (scaled by 1/TEMP) and negatives to f16.
//    Negatives zero-padded to N_PAD rows (pad contributes exp(0)=1, removed
//    as a constant in the loss kernel).
// ---------------------------------------------------------------------------
__global__ void gather_kernel(const float* __restrict__ feat,
                              const int*   __restrict__ anchor_idx,
                              const int*   __restrict__ neg_idx,
                              _Float16*    __restrict__ anchor_h,
                              _Float16*    __restrict__ neg_h) {
    const int e = blockIdx.x * blockDim.x + threadIdx.x;
    const int totalA = N_ANCH * CC;
    const int totalN = N_PAD * CC;
    if (e < totalA) {
        const int row = e >> 7, c = e & 127;
        anchor_h[e] = (_Float16)(gatherF(feat, anchor_idx[row], c) * INV_TEMP);
    } else if (e < totalA + totalN) {
        const int e2 = e - totalA;
        const int row = e2 >> 7, c = e2 & 127;
        float v = 0.0f;
        if (row < N_NEG) v = gatherF(feat, neg_idx[row], c);
        neg_h[e2] = (_Float16)v;
    }
}

// ---------------------------------------------------------------------------
// 3) pos_sum[c] = sum over 50000 gathered rows (mean folded in later)
// ---------------------------------------------------------------------------
__global__ void posmean_kernel(const float* __restrict__ feat,
                               const int*   __restrict__ pos_idx,
                               float*       __restrict__ pos_sum) {
    const int c = threadIdx.x;              // 128 threads per block
    float local = 0.0f;
    for (int r = blockIdx.x; r < N_POS; r += gridDim.x)
        local += gatherF(feat, pos_idx[r], c);
    atomicAdd(&pos_sum[c], local);
}

// ---------------------------------------------------------------------------
// 4) Main kernel: denom[a] = sum_n exp(anchor[a] . neg[n] / TEMP)
//    One wave per (16-row anchor tile, column chunk). K=128 via 4 chained
//    v_wmma_f32_16x16x32_f16; exp + rowsum fused in the epilogue; B tiles
//    software-pipelined (bnext loads issue before the bcur WMMA chain).
// ---------------------------------------------------------------------------
__device__ __forceinline__ v16h load_frag(const _Float16* __restrict__ rowBase,
                                          int kb, int hi) {
    // ISA 16-bit A layout (and B as N-major == B^T row-major):
    //   lane<16: K in {kb+0..7, kb+16..23};  lane>=16: K in {kb+8..15, kb+24..31}
    const v8h lo = *(const v8h*)(rowBase + kb + hi * 8);
    const v8h up = *(const v8h*)(rowBase + kb + 16 + hi * 8);
    return __builtin_shufflevector(lo, up, 0, 1, 2, 3, 4, 5, 6, 7,
                                          8, 9, 10, 11, 12, 13, 14, 15);
}

__device__ __forceinline__ void load_btile(const _Float16* __restrict__ neg_h,
                                           int t, int hl, int hi, v16h b[4]) {
    const _Float16* nrow = neg_h + (size_t)(t * 16 + hl) * CC;
#pragma unroll
    for (int kc = 0; kc < 4; ++kc)
        b[kc] = load_frag(nrow, kc * 32, hi);
}

__global__ __launch_bounds__(128) void
wmma_denom_kernel(const _Float16* __restrict__ anchor_h,
                  const _Float16* __restrict__ neg_h,
                  float*          __restrict__ denom) {
    const int lane  = threadIdx.x & 31;
    const int hl    = lane & 15;                       // position within half-wave
    const int hi    = lane >> 4;                       // 0 or 1
    // Wave coordinates are wave-uniform: force them scalar so loop control
    // stays on the SALU (no exec-mask loop handling).
    const int waveG   = __builtin_amdgcn_readfirstlane(blockIdx.x * 4 + (threadIdx.x >> 5));
    const int rowTile = waveG / N_CHUNK;               // 0..511
    const int chunk   = waveG % N_CHUNK;               // 0..15
    const int rowBase = rowTile * 16;

    // Load the wave's four A fragments once (full K=128 for its 16 anchors)
    const _Float16* arow = anchor_h + (size_t)(rowBase + hl) * CC;
    v16h afrag[4];
#pragma unroll
    for (int kc = 0; kc < 4; ++kc)
        afrag[kc] = load_frag(arow, kc * 32, hi);

    v8f acc = {};                                      // running exp-sums

    // Software pipeline: prime bcur, then each iteration issues bnext's loads
    // BEFORE the bcur WMMA chain + exp epilogue, hiding the load latency.
    v16h bcur[4], bnext[4];
    load_btile(neg_h, chunk, hl, hi, bcur);

    for (int t = chunk; t < N_TILES; t += N_CHUNK) {
        // Branch-free tail: clamp next-tile index (re-loads last tile once).
        const int tn = (t + N_CHUNK < N_TILES) ? (t + N_CHUNK) : t;

        // Speculative prefetch one chunk ahead of the pipelined load.
        __builtin_prefetch(neg_h + (size_t)((tn + N_CHUNK) * 16 + hl) * CC, 0, 3);

        load_btile(neg_h, tn, hl, hi, bnext);

        v8f c = {};
#pragma unroll
        for (int kc = 0; kc < 4; ++kc)
            c = __builtin_amdgcn_wmma_f32_16x16x32_f16(
                    false, afrag[kc], false, bcur[kc], (short)0, c, false, false);

        // Epilogue: exp + accumulate (pad columns contribute exp(0)=1,
        // removed as a constant later — no per-element masking).
#pragma unroll
        for (int r = 0; r < 8; ++r)
            acc[r] += __expf(c[r]);

#pragma unroll
        for (int kc = 0; kc < 4; ++kc)
            bcur[kc] = bnext[kc];
    }

    // Reduce across the 16 lanes of each half-wave (xor masks < 16 stay
    // inside each half). Lane 0 -> row rowBase+r, lane 16 -> rowBase+8+r.
#pragma unroll
    for (int r = 0; r < 8; ++r) {
        float s = acc[r];
        s += __shfl_xor(s, 1, 32);
        s += __shfl_xor(s, 2, 32);
        s += __shfl_xor(s, 4, 32);
        s += __shfl_xor(s, 8, 32);
        if (hl == 0)
            atomicAdd(&denom[rowBase + hi * 8 + r], s);
    }
}

// ---------------------------------------------------------------------------
// 5) loss[a] = ep / (ep + denom[a] - PAD_EXTRA),
//    ep = exp(anchor_scaled . pos_mean)   (anchor_h carries 1/TEMP)
// ---------------------------------------------------------------------------
__global__ void loss_kernel(const _Float16* __restrict__ anchor_h,
                            const float*    __restrict__ pos_sum,
                            const float*    __restrict__ denom,
                            float*          __restrict__ out) {
    __shared__ float pm[CC];
    if (threadIdx.x < CC) pm[threadIdx.x] = pos_sum[threadIdx.x] * (1.0f / (float)N_POS);
    __syncthreads();
    const int a = blockIdx.x * blockDim.x + threadIdx.x;
    if (a >= N_ANCH) return;
    const _Float16* row = anchor_h + (size_t)a * CC;
    float dot = 0.0f;
#pragma unroll 4
    for (int c = 0; c < CC; ++c) dot += (float)row[c] * pm[c];
    const float ep = __expf(dot);
    out[a] = ep / (ep + (denom[a] - PAD_EXTRA));
}

// ---------------------------------------------------------------------------
// Launcher
// ---------------------------------------------------------------------------
extern "C" void kernel_launch(void* const* d_in, const int* in_sizes, int n_in,
                              void* d_out, int out_size, void* d_ws, size_t ws_size,
                              hipStream_t stream) {
    const float* feat       = (const float*)d_in[0];
    const int*   anchor_idx = (const int*)  d_in[1];
    const int*   pos_idx    = (const int*)  d_in[2];
    const int*   neg_idx    = (const int*)  d_in[3];
    float*       out        = (float*)d_out;

    // Workspace layout (all 256B aligned)
    char* ws = (char*)d_ws;
    float*    pos_sum  = (float*)(ws + 0);                       // 128 f32
    float*    denom    = (float*)(ws + 1024);                    // 8192 f32
    _Float16* anchor_h = (_Float16*)(ws + 65536);                // 8192*128 f16 (2 MiB)
    _Float16* neg_h    = (_Float16*)(ws + 65536 + (size_t)N_ANCH * CC * 2); // 15008*128 f16

    // 1) zero accumulators
    init_kernel<<<(N_ANCH + 255) / 256, 256, 0, stream>>>(pos_sum, denom);

    // 2) gather + f16 pack (anchors scaled by 1/TEMP; negs zero-padded)
    {
        const int total = (N_ANCH + N_PAD) * CC;
        gather_kernel<<<(total + 255) / 256, 256, 0, stream>>>(
            feat, anchor_idx, neg_idx, anchor_h, neg_h);
    }

    // 3) positive-mean accumulation
    posmean_kernel<<<256, CC, 0, stream>>>(feat, pos_idx, pos_sum);

    // 4) fused WMMA GEMM + exp + rowsum: 512 row tiles * 16 chunks = 8192 waves
    {
        const int waves  = (N_ANCH / 16) * N_CHUNK;   // 8192
        const int blocks = waves / 4;                 // 128 threads = 4 waves/block
        wmma_denom_kernel<<<blocks, 128, 0, stream>>>(anchor_h, neg_h, denom);
    }

    // 5) final loss
    loss_kernel<<<(N_ANCH + 255) / 256, 256, 0, stream>>>(anchor_h, pos_sum, denom, out);
}
